// SelfAttention_86827058856619
// MI455X (gfx1250) — matrix operand order repaired and checked by reading, not compiled, not run on hardware
//
#include <hip/hip_runtime.h>
#include <hip/hip_bf16.h>

typedef __attribute__((ext_vector_type(16))) __bf16 bf16x16;
typedef __attribute__((ext_vector_type(8)))  __bf16 bf16x8;
typedef __attribute__((ext_vector_type(8)))  float  f32x8;

// ---------------------------------------------------------------------------
// Problem constants
// ---------------------------------------------------------------------------
#define BS   8
#define T    2048
#define C    128          // per-head dim
#define H    4
#define HC   (H*C)        // 512
#define N3   (3*HC)       // 1536
#define M_ROWS (BS*T)     // 16384
#define BH   (BS*H)       // 32
#define KT   32           // keys per attention iteration

// ---------------------------------------------------------------------------
// WMMA helpers (layouts per CDNA5 ISA 7.12.2)
// ---------------------------------------------------------------------------
__device__ __forceinline__ f32x8 wmma_bf16(bf16x16 a, bf16x16 b, f32x8 c) {
  return __builtin_amdgcn_wmma_f32_16x16x32_bf16(
      /*neg_a=*/false, a, /*neg_b=*/false, b,
      /*c_mod=*/(short)0, c, /*reuse_a=*/false, /*reuse_b=*/false);
}

// A-matrix 16x32 bf16 fragment: lane holds row m=l&15; elements 0..7 = K
// chunk at (l>>4)*16, elements 8..15 = K chunk at (l>>4)*16+8.
__device__ __forceinline__ bf16x16 make_frag(const __bf16* p0, const __bf16* p1) {
  bf16x8 lo = *(const bf16x8*)p0;
  bf16x8 hi = *(const bf16x8*)p1;
  bf16x16 r;
#pragma unroll
  for (int i = 0; i < 8; ++i) { r[i] = lo[i]; r[8 + i] = hi[i]; }
  return r;
}

__device__ __forceinline__ bf16x16 make_frag_f32(const float* p0, const float* p1) {
  bf16x16 r;
#pragma unroll
  for (int i = 0; i < 8; ++i) { r[i] = (__bf16)p0[i]; r[8 + i] = (__bf16)p1[i]; }
  return r;
}

// Async global->LDS copy of 16 bytes per lane (CDNA5 ASYNCcnt path).
__device__ __forceinline__ unsigned lds_off(const void* p) {
  // low 32 bits of a generic LDS address are the wave-relative LDS offset
  return (unsigned)(uintptr_t)p;
}
__device__ __forceinline__ void async_b128(unsigned lds_byte_off, const void* gptr) {
  asm volatile("global_load_async_to_lds_b128 %0, %1, off"
               :: "v"(lds_byte_off), "v"(gptr)
               : "memory");
}

// ---------------------------------------------------------------------------
// Prep: transpose + convert weights to bf16 [N, K] so B-fragments are
// contiguous 16-element loads.
// ---------------------------------------------------------------------------
__global__ void conv_wqkv_kernel(const float* __restrict__ w, __bf16* __restrict__ wT) {
  int idx = blockIdx.x * blockDim.x + threadIdx.x;     // over N3*C
  if (idx >= N3 * C) return;
  int n = idx / C, c = idx % C;
  wT[(size_t)n * C + c] = (__bf16)w[(size_t)c * N3 + n];
}

__global__ void conv_wproj_kernel(const float* __restrict__ w, __bf16* __restrict__ wT) {
  int idx = blockIdx.x * blockDim.x + threadIdx.x;     // over C*HC
  if (idx >= C * HC) return;
  int n = idx / HC, k = idx % HC;
  wT[(size_t)n * HC + k] = (__bf16)w[(size_t)k * C + n];
}

// ---------------------------------------------------------------------------
// QKV GEMM: [16384,128] x [128,1536] + bias, scatter into
//   q_bf [bh, t, c], k_bf [bh, t, c], vT_bf [bh, c, t]  (all bf16)
// One 16x64 strip per wave (4 N-tiles reuse each A fragment).
// ---------------------------------------------------------------------------
__global__ __launch_bounds__(128) void qkv_kernel(
    const float* __restrict__ x, const __bf16* __restrict__ wT,
    const float* __restrict__ bias,
    __bf16* __restrict__ qb, __bf16* __restrict__ kb, __bf16* __restrict__ vTb) {
  const int wave = threadIdx.x >> 5;
  const int lane = threadIdx.x & 31;
  const int mtile = blockIdx.x;                    // 0..1023
  const int ntbase = (blockIdx.y * 4 + wave) * 4;  // 0,4,...,92
  const int row0 = mtile * 16;
  const int m = lane & 15, hh = lane >> 4;

  const float* xr = x + (size_t)(row0 + m) * C;

  f32x8 acc[4];
#pragma unroll
  for (int nt = 0; nt < 4; ++nt) acc[nt] = (f32x8){};

#pragma unroll
  for (int f = 0; f < 4; ++f) {
    bf16x16 a = make_frag_f32(xr + f * 32 + hh * 16, xr + f * 32 + hh * 16 + 8);
#pragma unroll
    for (int nt = 0; nt < 4; ++nt) {
      const __bf16* wr = wT + (size_t)((ntbase + nt) * 16 + (lane & 15)) * C;
      bf16x16 b = *(const bf16x16*)(wr + f * 32 + hh * 16);
      acc[nt] = wmma_bf16(a, b, acc[nt]);
    }
  }

#pragma unroll
  for (int nt = 0; nt < 4; ++nt) {
    const int n = (ntbase + nt) * 16 + (lane & 15);
    const int which = n >> 9;            // 0=q 1=k 2=v
    const int h = (n >> 7) & (H - 1);
    const int c = n & (C - 1);
    const float bv = bias[n];
#pragma unroll
    for (int r = 0; r < 8; ++r) {
      const int row = row0 + r + hh * 8;           // b*T + t
      const int bidx = row >> 11;
      const int t = row & (T - 1);
      const int bh = bidx * H + h;
      __bf16 v = (__bf16)(acc[nt][r] + bv);
      if (which == 0)      qb [((size_t)bh * T + t) * C + c] = v;
      else if (which == 1) kb [((size_t)bh * T + t) * C + c] = v;
      else                 vTb[((size_t)bh * C + c) * T + t] = v;
    }
  }
}

// ---------------------------------------------------------------------------
// Flash attention: one block = 64 queries of one (b,h); K/V tiles (32 keys)
// are staged into double-buffered LDS via async global->LDS copies
// (ASYNCcnt), overlapped with WMMA compute. P transpose (D->A layout) goes
// through per-wave LDS. Output stored bf16 as attn[b, t, h, c].
// ---------------------------------------------------------------------------
__global__ __launch_bounds__(128) void attn_kernel(
    const __bf16* __restrict__ qb, const __bf16* __restrict__ kbuf,
    const __bf16* __restrict__ vTb, __bf16* __restrict__ ob) {
  __shared__ __align__(64) __bf16 Kt[2][KT][C];     // 2 x 8 KB
  __shared__ __align__(64) __bf16 Vt[2][C][KT];     // 2 x 8 KB
  __shared__ __align__(64) __bf16 Pt[4][16 * 32];   // 4 KB

  const int tid = threadIdx.x;
  const int wave = tid >> 5;
  const int lane = tid & 31;
  const int bh = blockIdx.x;                       // 0..31
  const int q0 = blockIdx.y * 64 + wave * 16;      // 0..2032
  const int m = lane & 15, hh = lane >> 4;

  const __bf16* Q  = qb   + (size_t)(bh * T + q0) * C;
  const __bf16* Kg = kbuf + (size_t)bh * T * C;    // [t][c]
  const __bf16* Vg = vTb  + (size_t)bh * C * T;    // [c][t]

  // Q A-fragments, kept in registers for the whole key loop
  bf16x16 qf[4];
#pragma unroll
  for (int f = 0; f < 4; ++f)
    qf[f] = make_frag(Q + (size_t)m * C + f * 32 + hh * 16,
                      Q + (size_t)m * C + f * 32 + hh * 16 + 8);

  f32x8 o[8];
  float mi[8], li[8];
#pragma unroll
  for (int ct = 0; ct < 8; ++ct) o[ct] = (f32x8){};
#pragma unroll
  for (int r = 0; r < 8; ++r) { mi[r] = -1e30f; li[r] = 0.f; }

  const float scale = 0.08838834764831845f;        // 1/sqrt(128)
  __bf16* pl = Pt[wave];

  // --- async tile fill: K tile = 8KB contiguous, V tile = 128 rows x 64B ---
  auto issue_tile = [&](int buf, int kb0) {
    const char* ks = (const char*)(Kg + (size_t)kb0 * C);
    unsigned kd = lds_off(&Kt[buf][0][0]);
#pragma unroll
    for (int j = 0; j < 4; ++j) {
      unsigned boff = (unsigned)(tid + j * 128) * 16u;   // 0..8176
      async_b128(kd + boff, ks + boff);
    }
    const char* vs = (const char*)(Vg + (size_t)tid * T + kb0);
    unsigned vd = lds_off(&Vt[buf][tid][0]);
#pragma unroll
    for (int j = 0; j < 4; ++j)
      async_b128(vd + j * 16, vs + j * 16);
  };

  const int NIT = T / KT;                          // 64
  issue_tile(0, 0);
  issue_tile(1, KT);

  for (int it = 0; it < NIT; ++it) {
    const int buf = it & 1;
    // oldest batch (8 async ops) complete; next batch may still be in flight
    if (it + 1 < NIT) asm volatile("s_wait_asynccnt 0x8" ::: "memory");
    else              asm volatile("s_wait_asynccnt 0x0" ::: "memory");
    __syncthreads();

    // ---- S = Q K^T for 2 tiles of 16 keys (B-fragments from LDS) ----
    f32x8 st[2];
#pragma unroll
    for (int t2 = 0; t2 < 2; ++t2) {
      f32x8 s = {};
      const __bf16* Krow = &Kt[buf][t2 * 16 + (lane & 15)][0];
#pragma unroll
      for (int f = 0; f < 4; ++f)
        s = wmma_bf16(qf[f], *(const bf16x16*)(Krow + f * 32 + hh * 16), s);
#pragma unroll
      for (int r = 0; r < 8; ++r) st[t2][r] = s[r] * scale;
    }

    // ---- online softmax (rows M = r + hh*8 live in a 16-lane half) ----
    float alpha[8];
#pragma unroll
    for (int r = 0; r < 8; ++r) {
      float mx = fmaxf(st[0][r], st[1][r]);
#pragma unroll
      for (int msk = 1; msk < 16; msk <<= 1) mx = fmaxf(mx, __shfl_xor(mx, msk, 32));
      const float mn = fmaxf(mi[r], mx);
      const float a  = __expf(mi[r] - mn);
      const float p0 = __expf(st[0][r] - mn);
      const float p1 = __expf(st[1][r] - mn);
      float rs = p0 + p1;
#pragma unroll
      for (int msk = 1; msk < 16; msk <<= 1) rs += __shfl_xor(rs, msk, 32);
      li[r] = li[r] * a + rs;
      mi[r] = mn;
      alpha[r] = a;
      st[0][r] = p0; st[1][r] = p1;
    }
#pragma unroll
    for (int ct = 0; ct < 8; ++ct)
#pragma unroll
      for (int r = 0; r < 8; ++r) o[ct][r] *= alpha[r];

    // ---- P: D-layout -> A-layout via per-wave LDS (bf16) ----
#pragma unroll
    for (int r = 0; r < 8; ++r) {
      pl[(r + hh * 8) * 32 +      (lane & 15)] = (__bf16)st[0][r];
      pl[(r + hh * 8) * 32 + 16 + (lane & 15)] = (__bf16)st[1][r];
    }
    asm volatile("s_wait_dscnt 0x0" ::: "memory");
    bf16x16 pf = make_frag(pl + m * 32 + hh * 16, pl + m * 32 + hh * 16 + 8);

    // ---- O += P V : 8 channel tiles, one WMMA each (K=32 keys) ----
#pragma unroll
    for (int ct = 0; ct < 8; ++ct) {
      const __bf16* Vp = &Vt[buf][ct * 16 + (lane & 15)][hh * 16];
      o[ct] = wmma_bf16(pf, *(const bf16x16*)Vp, o[ct]);
    }

    __syncthreads();   // everyone done reading buf before it is refilled
    if (it + 2 < NIT) issue_tile(buf, (it + 2) * KT);
  }

  // ---- epilogue: normalize, store attn[b, t, h, c] bf16 ----
  const int b = bh >> 2, h = bh & (H - 1);
#pragma unroll
  for (int r = 0; r < 8; ++r) {
    const float inv = 1.0f / li[r];
    const int t = q0 + r + hh * 8;
    __bf16* orow = ob + ((size_t)(b * T + t) * H + h) * C;
#pragma unroll
    for (int ct = 0; ct < 8; ++ct)
      orow[ct * 16 + (lane & 15)] = (__bf16)(o[ct][r] * inv);
  }
}

// ---------------------------------------------------------------------------
// Output projection: [16384, 512] x [512, 128] + bias -> f32 d_out
// Each wave computes 2 N-tiles, reusing every A fragment twice.
// ---------------------------------------------------------------------------
__global__ __launch_bounds__(128) void proj_kernel(
    const __bf16* __restrict__ a, const __bf16* __restrict__ wT,
    const float* __restrict__ bias, float* __restrict__ out) {
  const int wave = threadIdx.x >> 5;
  const int lane = threadIdx.x & 31;
  const int mtile = blockIdx.x;                  // 0..1023
  const int ntbase = wave * 2;                   // 0,2,4,6
  const int row0 = mtile * 16;
  const int m = lane & 15, hh = lane >> 4;

  const __bf16* ar = a + (size_t)(row0 + m) * HC;

  f32x8 acc[2];
  acc[0] = (f32x8){}; acc[1] = (f32x8){};
#pragma unroll
  for (int f = 0; f < 16; ++f) {
    bf16x16 af = make_frag(ar + f * 32 + hh * 16, ar + f * 32 + hh * 16 + 8);
#pragma unroll
    for (int nt = 0; nt < 2; ++nt) {
      const __bf16* br = wT + (size_t)((ntbase + nt) * 16 + (lane & 15)) * HC;
      bf16x16 bf = *(const bf16x16*)(br + f * 32 + hh * 16);
      acc[nt] = wmma_bf16(af, bf, acc[nt]);
    }
  }

#pragma unroll
  for (int nt = 0; nt < 2; ++nt) {
    const int n = (ntbase + nt) * 16 + (lane & 15);
    const float bv = bias[n];
#pragma unroll
    for (int r = 0; r < 8; ++r)
      out[(size_t)(row0 + r + hh * 8) * C + n] = acc[nt][r] + bv;
  }
}

// ---------------------------------------------------------------------------
// Launch
// ---------------------------------------------------------------------------
extern "C" void kernel_launch(void* const* d_in, const int* in_sizes, int n_in,
                              void* d_out, int out_size, void* d_ws, size_t ws_size,
                              hipStream_t stream) {
  const float* x      = (const float*)d_in[0];
  const float* w_qkv  = (const float*)d_in[1];
  const float* b_qkv  = (const float*)d_in[2];
  const float* w_proj = (const float*)d_in[3];
  const float* b_proj = (const float*)d_in[4];
  float* out = (float*)d_out;

  unsigned char* ws = (unsigned char*)d_ws;
  size_t off = 0;
  auto alloc = [&](size_t bytes) {
    void* p = ws + off;
    off = (off + bytes + 255) & ~(size_t)255;
    return p;
  };
  __bf16* wT_qkv = (__bf16*)alloc((size_t)N3 * C * 2);      // 384 KB
  __bf16* wT_prj = (__bf16*)alloc((size_t)C * HC * 2);      // 128 KB
  __bf16* q_bf   = (__bf16*)alloc((size_t)BH * T * C * 2);  // 16 MB
  __bf16* k_bf   = (__bf16*)alloc((size_t)BH * T * C * 2);  // 16 MB
  __bf16* vT_bf  = (__bf16*)alloc((size_t)BH * C * T * 2);  // 16 MB
  __bf16* at_bf  = (__bf16*)alloc((size_t)M_ROWS * HC * 2); // 16 MB
  (void)ws_size;

  conv_wqkv_kernel<<<(N3 * C + 255) / 256, 256, 0, stream>>>(w_qkv, wT_qkv);
  conv_wproj_kernel<<<(C * HC + 255) / 256, 256, 0, stream>>>(w_proj, wT_prj);

  qkv_kernel<<<dim3(M_ROWS / 16, (N3 / 64) / 4), 128, 0, stream>>>(
      x, wT_qkv, b_qkv, q_bf, k_bf, vT_bf);

  attn_kernel<<<dim3(BH, T / 64), 128, 0, stream>>>(q_bf, k_bf, vT_bf, at_bf);

  proj_kernel<<<dim3(M_ROWS / 16, 1), 128, 0, stream>>>(
      at_bf, wT_prj, b_proj, out);
}
